// TreeLSTM_Encoders_42477226557556
// MI455X (gfx1250) — matrix operand order repaired and checked by reading, not compiled, not run on hardware
//
#include <hip/hip_runtime.h>
#include <math.h>

#define DM 128
#define BTREES 64

typedef __attribute__((ext_vector_type(16))) __bf16        v16bf;
typedef __attribute__((ext_vector_type(8)))  float         v8f;
typedef __attribute__((ext_vector_type(4)))  unsigned int  u32x4;

union Frag {
    u32x4 q[2];
    v16bf v;
};
static_assert(sizeof(Frag) == 32, "frag size");

__device__ __forceinline__ unsigned short f2bf(float f) {
    union { float f; unsigned u; } v; v.f = f;
    unsigned r = (v.u + 0x7FFFu + ((v.u >> 16) & 1u)) >> 16;   // RNE
    return (unsigned short)r;
}

__device__ __forceinline__ float sigf(float x) {
    return 1.0f / (1.0f + __expf(-x));
}

// ---------------------------------------------------------------------------
// Pack a row-major f32 weight matrix W[K][NW] into a contiguous-k (k = 16*h + e) WMMA bf16
// B-matrix lane layout:  per (k-chunk kb of 32, n-block nb of 16):
//   lane = (n%16) + 16*((k%32) >= 16),  element e = (k%32) & 15
//   P[(((kb*(NW/16))+nb)*32 + lane)*16 + e]
// so each wave's B fragment is 32 contiguous bytes per lane (two b128 loads).
// ---------------------------------------------------------------------------
__global__ void pack_w_kernel(const float* __restrict__ W,
                              unsigned short* __restrict__ P,
                              int K, int NW) {
    int idx = blockIdx.x * blockDim.x + threadIdx.x;
    if (idx >= K * NW) return;
    int k = idx / NW, n = idx % NW;
    int kb = k >> 5, kr = k & 31;
    int nb = n >> 4;
    int lane = (n & 15) + ((kr >= 16) ? 16 : 0);
    int e = kr & 15;
    long off = (((long)(kb * (NW >> 4) + nb)) * 32 + lane) * 16 + e;
    P[off] = f2bf(W[(long)k * NW + n]);
}

// ---------------------------------------------------------------------------
// Leaves: h = emb[token] * min(1, 1/max(||e||,1e-7)), c = 0.
// One block of 128 threads per leaf node.
// ---------------------------------------------------------------------------
__global__ void leaf_kernel(const int* __restrict__ tokens,
                            const float* __restrict__ emb,
                            unsigned short* __restrict__ hbf,
                            float* __restrict__ cbuf,
                            float* __restrict__ out,
                            int C0) {
    int node = blockIdx.x;
    int d = threadIdx.x;
    __shared__ float red[DM];
    int tok = tokens[node];
    float e = emb[(long)tok * DM + d];
    red[d] = e * e;
    __syncthreads();
    for (int s = DM / 2; s > 0; s >>= 1) {
        if (d < s) red[d] += red[d + s];
        __syncthreads();
    }
    float nrm = sqrtf(red[0]);
    float scale = fminf(1.0f, 1.0f / fmaxf(nrm, 1e-7f));
    float h = e * scale;
    hbf[(long)node * DM + d] = f2bf(h);
    cbuf[(long)node * DM + d] = 0.0f;
    int b = node / C0, j = node % C0;
    out[((long)b * 1024 + j) * DM + d] = h;
}

// ---------------------------------------------------------------------------
// One tree level, fully fused and compile-time specialized:
//   gather children h (bf16) -> bf16 WMMA GEMM against packed W ->
//   all NGATES gate slices in-register -> LSTM cell update ->
//   store c (f32), h (bf16 for next level), h (f32 into output slot).
//
// Wave tile: 16 nodes x 16 dims of DM; NGATES v8f accumulators.
// BINARY=1: K=256 (left|right), 5 gates.  BINARY=0: K=128, 4 gates.
// ---------------------------------------------------------------------------
template <int BINARY, int NGATES, int KCHUNKS>
__global__ void tree_level_kernel(unsigned short* __restrict__ hbf,  // children in, this level out
                                  float* __restrict__ cbuf,
                                  const unsigned short* __restrict__ Wp,
                                  const float* __restrict__ bias,
                                  const int* __restrict__ left_idx,
                                  const int* __restrict__ right_idx,
                                  float* __restrict__ out,
                                  int sNode, int M, int clShift, int offIntra) {
    const int lane = threadIdx.x & 31;
    const int wid  = threadIdx.x >> 5;
    const int t    = blockIdx.x * (blockDim.x >> 5) + wid;
    const int mt   = t >> 3;        // node tile (16 nodes)
    const int nt   = t & 7;         // dim tile  (16 of DM=128)
    if (mt * 16 >= M) return;

    // Children of the A-matrix row held by this lane (A rows = lane % 16).
    const int rowNode = sNode + mt * 16 + (lane & 15);
    const long lidx = left_idx[rowNode];
    const long ridx = BINARY ? (long)right_idx[rowNode] : lidx;

    const v8f vzero = {0.f, 0.f, 0.f, 0.f, 0.f, 0.f, 0.f, 0.f};
    v8f acc[NGATES];
#pragma unroll
    for (int g = 0; g < NGATES; ++g) acc[g] = vzero;

    constexpr int nbStride = NGATES * 8;      // n-blocks per k-chunk (NW/16)

#pragma unroll
    for (int kb = 0; kb < KCHUNKS; ++kb) {
        // 16-bit A 16x32 lane layout (matches packed B): lane<16 holds k in
        // [kb*32+0..15]; lane>=16 holds [+16..31].
        const int kbase = kb * 32 + ((lane >= 16) ? 16 : 0);
        Frag a;
        {
            const int k1 = kbase;             // 8 contiguous k
            const int k2 = kbase + 8;         // 8 contiguous k
            // Child selection is compile-time per kb (kb<4 -> left, else right).
            const long s1 = (kb * 32 < DM) ? lidx : ridx;
            const long s2 = (kb * 32 + 16 < DM + 16 && kb * 32 >= DM) ? ridx
                             : ((k2 < DM) ? lidx : ridx);
            a.q[0] = *(const u32x4*)(hbf + s1 * DM + (k1 & (DM - 1)));
            a.q[1] = *(const u32x4*)(hbf + s2 * DM + (k2 & (DM - 1)));
        }
        const unsigned short* wkb = Wp + ((long)kb * nbStride) * 32 * 16;
#pragma unroll
        for (int g = 0; g < NGATES; ++g) {
            const unsigned short* bp = wkb + ((long)(g * 8 + nt) * 32 + lane) * 16;
            Frag b;
            b.q[0] = *(const u32x4*)(bp);
            b.q[1] = *(const u32x4*)(bp + 8);
            acc[g] = __builtin_amdgcn_wmma_f32_16x16x32_bf16(
                false, a.v, false, b.v, (short)0, acc[g], false, false);
        }
    }

    // ------------------------- fused LSTM epilogue -------------------------
    const int dcol = nt * 16 + (lane & 15);
    float bb[NGATES];
#pragma unroll
    for (int g = 0; g < NGATES; ++g) bb[g] = bias[g * DM + dcol];

#pragma unroll
    for (int r = 0; r < 8; ++r) {
        // f32 C/D layout: VGPR r, lanes 0-15 -> M=r, lanes 16-31 -> M=8+r.
        const int m = mt * 16 + r + ((lane >= 16) ? 8 : 0);
        const long node = sNode + m;
        const long li = left_idx[node];
        const float lc = cbuf[li * DM + dcol];
        const float iG = acc[0][r] + bb[0];
        const float oG = acc[1][r] + bb[1];
        const float uG = acc[2][r] + bb[2];
        float c;
        if (BINARY) {
            const long ri = right_idx[node];
            const float rc = cbuf[ri * DM + dcol];
            const float flG = acc[3][r] + bb[3];
            const float frG = acc[NGATES - 1][r] + bb[NGATES - 1];
            c = sigf(iG) * tanhf(uG) + sigf(flG) * lc + sigf(frG) * rc;
        } else {
            const float fG = acc[3][r] + bb[3];
            c = sigf(iG) * tanhf(uG) + sigf(fG) * lc;
        }
        const float h = sigf(oG) * tanhf(c);
        cbuf[node * DM + dcol] = c;
        hbf[node * DM + dcol] = f2bf(h);
        const int bT = m >> clShift;              // Cl is a power of two
        const int j  = m & ((1 << clShift) - 1);
        out[((long)bT * 1024 + offIntra + j) * DM + dcol] = h;
    }
}

// ---------------------------------------------------------------------------
extern "C" void kernel_launch(void* const* d_in, const int* in_sizes, int n_in,
                              void* d_out, int out_size, void* d_ws, size_t ws_size,
                              hipStream_t stream) {
    const int*   tokens    = (const int*)d_in[0];
    const int*   left_idx  = (const int*)d_in[1];
    const int*   right_idx = (const int*)d_in[2];
    const float* emb       = (const float*)d_in[3];
    const float* W_un      = (const float*)d_in[4];
    const float* b_un      = (const float*)d_in[5];
    const float* W_bin     = (const float*)d_in[6];
    const float* b_bin     = (const float*)d_in[7];
    float* out = (float*)d_out;

    static const int COUNTS[11] = {512, 256, 128, 64, 32, 16, 8, 4, 2, 1, 1};
    int G[12];  G[0] = 0;
    for (int l = 0; l < 11; ++l) G[l + 1] = G[l] + BTREES * COUNTS[l];
    int OFF[11]; OFF[0] = 0;
    for (int l = 1; l < 11; ++l) OFF[l] = OFF[l - 1] + COUNTS[l - 1];
    const long N = G[11];   // 65536 nodes

    // Workspace layout
    char* ws = (char*)d_ws;
    unsigned short* hbf   = (unsigned short*)ws;                       // N*DM*2  = 16 MB
    float*          cbuf  = (float*)(ws + (size_t)N * DM * 2);         // N*DM*4  = 32 MB
    unsigned short* WbinP = (unsigned short*)(ws + (size_t)N * DM * 6);
    unsigned short* WunP  = WbinP + 256 * 640;

    // Pack weights into WMMA bf16 B-matrix layout (every call; deterministic).
    pack_w_kernel<<<(256 * 640 + 255) / 256, 256, 0, stream>>>(W_bin, WbinP, 256, 640);
    pack_w_kernel<<<(128 * 512 + 255) / 256, 256, 0, stream>>>(W_un,  WunP,  128, 512);

    // Level 0: leaves.
    leaf_kernel<<<G[1], DM, 0, stream>>>(tokens, emb, hbf, cbuf, out, COUNTS[0]);

    // Levels 1..9: binary (K=256, 5 gates).  Level 10: unary (K=128, 4 gates).
    for (int l = 1; l <= 10; ++l) {
        const int M = BTREES * COUNTS[l];
        int clShift = 0;
        while ((1 << clShift) < COUNTS[l]) ++clShift;   // log2(COUNTS[l])
        const int blocks = M / 16;                      // 8 waves/block, 1 tile/wave
        if (l < 10) {
            tree_level_kernel<1, 5, 8><<<blocks, 256, 0, stream>>>(
                hbf, cbuf, WbinP, b_bin, left_idx, right_idx, out,
                G[l], M, clShift, OFF[l]);
        } else {
            tree_level_kernel<0, 4, 4><<<blocks, 256, 0, stream>>>(
                hbf, cbuf, WunP, b_un, left_idx, right_idx, out,
                G[l], M, clShift, OFF[l]);
        }
    }
}
